// MyModel_84911503442110
// MI455X (gfx1250) — matrix operand order repaired and checked by reading, not compiled, not run on hardware
//
#include <hip/hip_runtime.h>
#include <hip/hip_bf16.h>

#define BATCH 64
#define SLEN 32
#define TLEN 32
#define HID 512
#define VOUTN 32000

typedef __attribute__((ext_vector_type(8)))  float  v8f;
typedef __attribute__((ext_vector_type(16))) __bf16 v16bf;
typedef __attribute__((ext_vector_type(8)))  __bf16 v8bf;

__device__ __forceinline__ unsigned short f2bf(float f) {
  unsigned int u = __float_as_uint(f);
  u += 0x7FFFu + ((u >> 16) & 1u);          // round-to-nearest-even
  return (unsigned short)(u >> 16);
}
__device__ __forceinline__ float sigm(float x) { return 1.0f / (1.0f + __expf(-x)); }

// ---------------------------------------------------------------------------
// Generic bf16 WMMA GEMM:  C[m][n] = sum_k A[m][k]*W[n][k] (+bias[n]) (+Cin[m][n])
// One wave -> 16(M) x 64(N) tile via 4 accumulators reusing the A fragment.
// Optional output-row remap: m=(t*remapB+b) -> row (b*remapT+t) (for [B,T,V] logits).
// ---------------------------------------------------------------------------
__global__ __launch_bounds__(128) void k_gemm(
    const unsigned short* __restrict__ A, int lda,
    const unsigned short* __restrict__ W, int ldw,
    float* __restrict__ C, int ldc,
    const float* __restrict__ bias,
    const float* __restrict__ Cin, int ldcin,
    int M, int N, int K, int remapB, int remapT)
{
  const int lane = threadIdx.x;        // 0..31 (wave32)
  const int half = lane >> 4;
  const int nn   = lane & 15;
  const int n0   = blockIdx.x * 64;
  const int m0   = (blockIdx.y * blockDim.y + threadIdx.y) * 16;
  if (m0 >= M) return;                 // wave-uniform; EXEC stays all-ones otherwise

  // A fragment base: lane r holds row m0+r; elems 0..15 = K half*16+0..15
  const unsigned short* Arow = A + (size_t)(m0 + nn) * lda + half * 16;
  // B fragment base: lane holds column n; elems 0..15 = K half*16+0..15 (contiguous in W row n)
  const unsigned short* W0 = W + (size_t)(n0 +      nn) * ldw + half * 16;
  const unsigned short* W1 = W + (size_t)(n0 + 16 + nn) * ldw + half * 16;
  const unsigned short* W2 = W + (size_t)(n0 + 32 + nn) * ldw + half * 16;
  const unsigned short* W3 = W + (size_t)(n0 + 48 + nn) * ldw + half * 16;

  v8f acc0 = {}, acc1 = {}, acc2 = {}, acc3 = {};
  for (int k0 = 0; k0 < K; k0 += 32) {
    v8bf alo = *(const v8bf*)(Arow + k0);
    v8bf ahi = *(const v8bf*)(Arow + k0 + 8);
    v16bf a  = __builtin_shufflevector(alo, ahi, 0,1,2,3,4,5,6,7,8,9,10,11,12,13,14,15);
    v16bf b0 = *(const v16bf*)(W0 + k0);
    v16bf b1 = *(const v16bf*)(W1 + k0);
    v16bf b2 = *(const v16bf*)(W2 + k0);
    v16bf b3 = *(const v16bf*)(W3 + k0);
    acc0 = __builtin_amdgcn_wmma_f32_16x16x32_bf16(false, a, false, b0, (short)0, acc0, false, false);
    acc1 = __builtin_amdgcn_wmma_f32_16x16x32_bf16(false, a, false, b1, (short)0, acc1, false, false);
    acc2 = __builtin_amdgcn_wmma_f32_16x16x32_bf16(false, a, false, b2, (short)0, acc2, false, false);
    acc3 = __builtin_amdgcn_wmma_f32_16x16x32_bf16(false, a, false, b3, (short)0, acc3, false, false);
  }

  float b0v = 0.f, b1v = 0.f, b2v = 0.f, b3v = 0.f;
  if (bias) { b0v = bias[n0+nn]; b1v = bias[n0+16+nn]; b2v = bias[n0+32+nn]; b3v = bias[n0+48+nn]; }

#pragma unroll
  for (int d = 0; d < 8; d++) {
    int m = m0 + half * 8 + d;         // C/D layout: VGPR d -> row d (+8 for upper half)
    float c0 = acc0[d] + b0v, c1 = acc1[d] + b1v, c2 = acc2[d] + b2v, c3 = acc3[d] + b3v;
    if (Cin) {
      const float* ci = Cin + (size_t)m * ldcin + n0;
      c0 += ci[nn]; c1 += ci[16+nn]; c2 += ci[32+nn]; c3 += ci[48+nn];
    }
    size_t orow = (remapB > 0)
      ? (size_t)((m % remapB) * remapT + m / remapB) * (size_t)ldc
      : (size_t)m * (size_t)ldc;
    float* co = C + orow + n0;
    co[nn] = c0; co[16+nn] = c1; co[32+nn] = c2; co[48+nn] = c3;
  }
}

// ---------------------------------------------------------------------------
// Elementwise / helper kernels
// ---------------------------------------------------------------------------
__global__ void k_cvt(const float* __restrict__ in, unsigned short* __restrict__ out, long n) {
  long i = (long)blockIdx.x * blockDim.x + threadIdx.x;
  if (i < n) out[i] = f2bf(in[i]);
}

__global__ void k_biassum(const float* a, const float* b, float* o, int n) {
  int i = blockIdx.x * blockDim.x + threadIdx.x;
  if (i < n) o[i] = a[i] + b[i];
}

// Wcat[n] = [ Wih_d[n][512:1536] | Whh_d[n][0:512] ]  (bf16, ld=1536)
__global__ void k_build_wcat(const float* __restrict__ Wih_d, const float* __restrict__ Whh_d,
                             unsigned short* __restrict__ out) {
  long i = (long)blockIdx.x * blockDim.x + threadIdx.x;
  if (i >= (long)2048 * 1536) return;
  int n = (int)(i / 1536), j = (int)(i % 1536);
  float v = (j < 1024) ? Wih_d[(size_t)n * 1536 + 512 + j] : Whh_d[(size_t)n * 512 + (j - 1024)];
  out[i] = f2bf(v);
}

// Xenc[(s*B+b)*H+j] = bf16(emb_enc[input[b][s]][j])
__global__ void k_gather_enc(const int* __restrict__ tok, const float* __restrict__ emb,
                             unsigned short* __restrict__ out) {
  long i = (long)blockIdx.x * blockDim.x + threadIdx.x;
  if (i >= (long)SLEN * BATCH * HID) return;
  int j = (int)(i % HID); long sb = i / HID;
  int b = (int)(sb % BATCH), s = (int)(sb / BATCH);
  int t = tok[b * SLEN + s];
  out[i] = f2bf(emb[(size_t)t * HID + j]);
}

// EmbDec[(t*B+b)*H+j] = bf16(emb_dec[dec_in[t][b]][j]); dec_in = [SOS, target[:, :-1]]
__global__ void k_gather_dec(const int* __restrict__ target, const float* __restrict__ emb,
                             unsigned short* __restrict__ out) {
  long i = (long)blockIdx.x * blockDim.x + threadIdx.x;
  if (i >= (long)TLEN * BATCH * HID) return;
  int j = (int)(i % HID); long tb = i / HID;
  int b = (int)(tb % BATCH), t = (int)(tb / BATCH);
  int tk = (t == 0) ? 1 : target[b * TLEN + (t - 1)];
  out[i] = f2bf(emb[(size_t)tk * HID + j]);
}

// LSTM gate nonlinearity (encoder): g=[B,4H] (i,f,g,o); updates h,c; writes bf16 h and enc_out
__global__ void k_enc_gate(const float* __restrict__ g, float* __restrict__ hst, float* __restrict__ cst,
                           unsigned short* __restrict__ hbf, float* __restrict__ enc_f,
                           unsigned short* __restrict__ enc_bf, int s, int dir) {
  int i = blockIdx.x * blockDim.x + threadIdx.x;
  if (i >= BATCH * HID) return;
  int b = i >> 9, j = i & 511;
  const float* gr = g + (size_t)b * 4 * HID;
  float c = sigm(gr[HID + j]) * cst[i] + sigm(gr[j]) * tanhf(gr[2 * HID + j]);
  float h = sigm(gr[3 * HID + j]) * tanhf(c);
  cst[i] = c; hst[i] = h; hbf[i] = f2bf(h);
  size_t eo = ((size_t)b * SLEN + s) * (2 * HID) + (size_t)dir * HID + j;
  enc_f[eo] = h; enc_bf[eo] = f2bf(h);
}

// Decoder gate: also writes h(bf16) into xcat[:,1024:1536] and Hdec[t] for the logits GEMM
__global__ void k_dec_gate(const float* __restrict__ g, float* __restrict__ hst, float* __restrict__ cst,
                           unsigned short* __restrict__ xcat, unsigned short* __restrict__ Hdec, int t) {
  int i = blockIdx.x * blockDim.x + threadIdx.x;
  if (i >= BATCH * HID) return;
  int b = i >> 9, j = i & 511;
  const float* gr = g + (size_t)b * 4 * HID;
  float c = sigm(gr[HID + j]) * cst[i] + sigm(gr[j]) * tanhf(gr[2 * HID + j]);
  float h = sigm(gr[3 * HID + j]) * tanhf(c);
  cst[i] = c; hst[i] = h;
  xcat[(size_t)b * 1536 + 1024 + j] = f2bf(h);
  Hdec[((size_t)t * BATCH + b) * HID + j] = f2bf(h);
}

// hcat=[hf|hb] bf16, ccat=[cf|cb] bf16
__global__ void k_cat_states(const float* hf, const float* hb, const float* cf, const float* cb,
                             unsigned short* hcat, unsigned short* ccat) {
  int i = blockIdx.x * blockDim.x + threadIdx.x;
  if (i >= BATCH * 2 * HID) return;
  int b = i >> 10, j = i & 1023;
  float hv = (j < HID) ? hf[(b << 9) + j] : hb[(b << 9) + j - HID];
  float cv = (j < HID) ? cf[(b << 9) + j] : cb[(b << 9) + j - HID];
  hcat[i] = f2bf(hv); ccat[i] = f2bf(cv);
}

__global__ void k_h2xcat(const float* h0, unsigned short* xcat) {
  int i = blockIdx.x * blockDim.x + threadIdx.x;
  if (i >= BATCH * HID) return;
  int b = i >> 9, j = i & 511;
  xcat[(size_t)b * 1536 + 1024 + j] = f2bf(h0[i]);
}

// scores[b][s] = sum_h tanh(q[b][h] + ua[b*S+s][h]) * Va[h] + bva  (one wave per (b,s))
__global__ void k_attn_scores(const float* __restrict__ q, const float* __restrict__ ua,
                              const float* __restrict__ Va, const float* __restrict__ bva,
                              float* __restrict__ scores) {
  int row  = blockIdx.x * blockDim.y + threadIdx.y;   // 0..B*S-1
  int lane = threadIdx.x;
  int b = row / SLEN;
  const float* qr = q + (size_t)b * HID;
  const float* ur = ua + (size_t)row * HID;
  float acc = 0.f;
  for (int h = lane; h < HID; h += 32) acc += tanhf(qr[h] + ur[h]) * Va[h];
#pragma unroll
  for (int off = 16; off; off >>= 1) acc += __shfl_xor(acc, off, 32);
  if (lane == 0) scores[row] = acc + bva[0];
}

// softmax over S=32 (wave0) + ctx = w @ enc_out -> xcat[:, :1024] (bf16); writes attn to d_out
__global__ void k_attn_ctx(const float* __restrict__ scores, const float* __restrict__ enc_f,
                           unsigned short* __restrict__ xcat, float* __restrict__ attn, int t) {
  __shared__ float sw[SLEN];
  int b = blockIdx.x, tid = threadIdx.x;
  if (tid < 32) {
    float sc = scores[b * SLEN + tid];
    float mx = sc;
#pragma unroll
    for (int off = 16; off; off >>= 1) mx = fmaxf(mx, __shfl_xor(mx, off, 32));
    float e = __expf(sc - mx);
    float sum = e;
#pragma unroll
    for (int off = 16; off; off >>= 1) sum += __shfl_xor(sum, off, 32);
    float w = e / sum;
    sw[tid] = w;
    attn[((size_t)b * TLEN + t) * SLEN + tid] = w;
  }
  __syncthreads();
  for (int d = tid; d < 2 * HID; d += blockDim.x) {
    const float* eb = enc_f + (size_t)b * SLEN * 2 * HID + d;
    float acc = 0.f;
#pragma unroll 4
    for (int s = 0; s < SLEN; s++) acc += sw[s] * eb[(size_t)s * 2 * HID];
    xcat[(size_t)b * 1536 + d] = f2bf(acc);
  }
}

// in-place log-softmax over V per row (block per row)
__global__ __launch_bounds__(256) void k_logsoftmax(float* __restrict__ out, int V) {
  __shared__ float red[256];
  float* p = out + (size_t)blockIdx.x * V;
  int tid = threadIdx.x;
  float m = -3.4e38f;
  for (int i = tid; i < V; i += 256) m = fmaxf(m, p[i]);
  red[tid] = m; __syncthreads();
  for (int s = 128; s; s >>= 1) { if (tid < s) red[tid] = fmaxf(red[tid], red[tid + s]); __syncthreads(); }
  m = red[0]; __syncthreads();
  float sum = 0.f;
  for (int i = tid; i < V; i += 256) sum += __expf(p[i] - m);
  red[tid] = sum; __syncthreads();
  for (int s = 128; s; s >>= 1) { if (tid < s) red[tid] += red[tid + s]; __syncthreads(); }
  float lse = m + __logf(red[0]); __syncthreads();
  for (int i = tid; i < V; i += 256) p[i] = p[i] - lse;
}

__global__ void k_copy_states(const float* hd, const float* cd, float* outH, float* outC) {
  int i = blockIdx.x * blockDim.x + threadIdx.x;
  if (i >= BATCH * HID) return;
  outH[i] = hd[i]; outC[i] = cd[i];
}

// ---------------------------------------------------------------------------
extern "C" void kernel_launch(void* const* d_in, const int* in_sizes, int n_in,
                              void* d_out, int out_size, void* d_ws, size_t ws_size,
                              hipStream_t stream) {
  const int*   input   = (const int*)d_in[0];
  const int*   target  = (const int*)d_in[1];
  const float* emb_enc = (const float*)d_in[2];
  const float* Wih_f = (const float*)d_in[3];  const float* Whh_f = (const float*)d_in[4];
  const float* bih_f = (const float*)d_in[5];  const float* bhh_f = (const float*)d_in[6];
  const float* Wih_b = (const float*)d_in[7];  const float* Whh_b = (const float*)d_in[8];
  const float* bih_b = (const float*)d_in[9];  const float* bhh_b = (const float*)d_in[10];
  const float* Wa  = (const float*)d_in[11];   const float* ba  = (const float*)d_in[12];
  const float* Ua  = (const float*)d_in[13];   const float* bua = (const float*)d_in[14];
  const float* Va  = (const float*)d_in[15];   const float* bva = (const float*)d_in[16];
  const float* emb_dec = (const float*)d_in[17];
  const float* Wih_d = (const float*)d_in[18]; const float* Whh_d = (const float*)d_in[19];
  const float* bih_d = (const float*)d_in[20]; const float* bhh_d = (const float*)d_in[21];
  const float* Wout = (const float*)d_in[22];  const float* bout = (const float*)d_in[23];
  const float* Wrh = (const float*)d_in[24];   const float* brh = (const float*)d_in[25];
  const float* Wrc = (const float*)d_in[26];   const float* brc = (const float*)d_in[27];

  const int B = BATCH, S = SLEN, T = TLEN, H = HID, V = VOUTN;

  size_t off = 0;
  auto alloc = [&](size_t bytes) -> void* {
    off = (off + 255) & ~(size_t)255;
    void* p = (char*)d_ws + off; off += bytes; return p;
  };
  typedef unsigned short us;
  us* Wihf_bf = (us*)alloc((size_t)4*H*H * 2);
  us* Whhf_bf = (us*)alloc((size_t)4*H*H * 2);
  us* Wihb_bf = (us*)alloc((size_t)4*H*H * 2);
  us* Whhb_bf = (us*)alloc((size_t)4*H*H * 2);
  us* Wabf    = (us*)alloc((size_t)H*H * 2);
  us* Uabf    = (us*)alloc((size_t)H*2*H * 2);
  us* Wrhbf   = (us*)alloc((size_t)H*2*H * 2);
  us* Wrcbf   = (us*)alloc((size_t)H*2*H * 2);
  us* Wihdbf  = (us*)alloc((size_t)4*H*3*H * 2);
  us* Wcatbf  = (us*)alloc((size_t)4*H*3*H * 2);
  us* Woutbf  = (us*)alloc((size_t)V*H * 2);
  float* bsum_f = (float*)alloc((size_t)4*H * 4);
  float* bsum_b = (float*)alloc((size_t)4*H * 4);
  float* bsum_d = (float*)alloc((size_t)4*H * 4);
  us* Xenc   = (us*)alloc((size_t)S*B*H * 2);
  us* EmbDec = (us*)alloc((size_t)T*B*H * 2);
  float* Gxf  = (float*)alloc((size_t)S*B*4*H * 4);
  float* Gxb  = (float*)alloc((size_t)S*B*4*H * 4);
  float* Gemb = (float*)alloc((size_t)T*B*4*H * 4);
  float* enc_f  = (float*)alloc((size_t)B*S*2*H * 4);
  us*    enc_bf = (us*)alloc((size_t)B*S*2*H * 2);
  float* ua_keys = (float*)alloc((size_t)B*S*H * 4);
  float* hf = (float*)alloc((size_t)B*H * 4); float* cf = (float*)alloc((size_t)B*H * 4);
  float* hb = (float*)alloc((size_t)B*H * 4); float* cb = (float*)alloc((size_t)B*H * 4);
  float* hd = (float*)alloc((size_t)B*H * 4); float* cd = (float*)alloc((size_t)B*H * 4);
  us* Hbf_f = (us*)alloc((size_t)B*H * 2);
  us* Hbf_b = (us*)alloc((size_t)B*H * 2);
  float* gbuf_f = (float*)alloc((size_t)B*4*H * 4);
  float* gbuf_b = (float*)alloc((size_t)B*4*H * 4);
  float* qbuf   = (float*)alloc((size_t)B*H * 4);
  float* scores = (float*)alloc((size_t)B*S * 4);
  us* hcat = (us*)alloc((size_t)B*2*H * 2);
  us* ccat = (us*)alloc((size_t)B*2*H * 2);
  us* xcat = (us*)alloc((size_t)B*3*H * 2);
  us* Hdec = (us*)alloc((size_t)T*B*H * 2);

  float* out = (float*)d_out;
  const size_t OFF_HT = (size_t)B * T * V;
  const size_t OFF_CT = OFF_HT + (size_t)B * H;
  const size_t OFF_AT = OFF_CT + (size_t)B * H;

  auto cvt = [&](const float* src, us* dst, long n) {
    k_cvt<<<dim3((unsigned)((n + 255) / 256)), dim3(256), 0, stream>>>(src, dst, n);
  };
  auto gemm = [&](const us* A, int lda, const us* W, int ldw, float* C, int ldc,
                  const float* bias, const float* Cin, int ldcin,
                  int M, int N, int K, int remapB, int remapT) {
    dim3 grid(N / 64, (M + 63) / 64), block(32, 4);
    k_gemm<<<grid, block, 0, stream>>>(A, lda, W, ldw, C, ldc, bias, Cin, ldcin, M, N, K, remapB, remapT);
  };

  // ---- one-time conversions & precompute ----
  cvt(Wih_f, Wihf_bf, (long)4*H*H);  cvt(Whh_f, Whhf_bf, (long)4*H*H);
  cvt(Wih_b, Wihb_bf, (long)4*H*H);  cvt(Whh_b, Whhb_bf, (long)4*H*H);
  cvt(Wa, Wabf, (long)H*H);          cvt(Ua, Uabf, (long)H*2*H);
  cvt(Wrh, Wrhbf, (long)H*2*H);      cvt(Wrc, Wrcbf, (long)H*2*H);
  cvt(Wih_d, Wihdbf, (long)4*H*3*H); cvt(Wout, Woutbf, (long)V*H);
  k_build_wcat<<<dim3((unsigned)(((long)4*H*3*H + 255) / 256)), dim3(256), 0, stream>>>(Wih_d, Whh_d, Wcatbf);
  k_biassum<<<dim3(8), dim3(256), 0, stream>>>(bih_f, bhh_f, bsum_f, 4*H);
  k_biassum<<<dim3(8), dim3(256), 0, stream>>>(bih_b, bhh_b, bsum_b, 4*H);
  k_biassum<<<dim3(8), dim3(256), 0, stream>>>(bih_d, bhh_d, bsum_d, 4*H);
  k_gather_enc<<<dim3((S*B*H + 255) / 256), dim3(256), 0, stream>>>(input, emb_enc, Xenc);
  k_gather_dec<<<dim3((T*B*H + 255) / 256), dim3(256), 0, stream>>>(target, emb_dec, EmbDec);

  // hoisted input-side GEMMs (all timesteps at once)
  gemm(Xenc,   H, Wihf_bf, H,   Gxf,  4*H, bsum_f, nullptr, 0, S*B, 4*H, H, 0, 0);
  gemm(Xenc,   H, Wihb_bf, H,   Gxb,  4*H, bsum_b, nullptr, 0, S*B, 4*H, H, 0, 0);
  gemm(EmbDec, H, Wihdbf,  3*H, Gemb, 4*H, bsum_d, nullptr, 0, T*B, 4*H, H, 0, 0);

  // ---- encoder recurrence (fwd + bwd per iteration) ----
  hipMemsetAsync(cf, 0, (size_t)B*H*4, stream);
  hipMemsetAsync(cb, 0, (size_t)B*H*4, stream);
  hipMemsetAsync(Hbf_f, 0, (size_t)B*H*2, stream);
  hipMemsetAsync(Hbf_b, 0, (size_t)B*H*2, stream);
  for (int i = 0; i < S; i++) {
    gemm(Hbf_f, H, Whhf_bf, H, gbuf_f, 4*H, nullptr, Gxf + (size_t)i*B*4*H, 4*H, B, 4*H, H, 0, 0);
    k_enc_gate<<<dim3((B*H + 255)/256), dim3(256), 0, stream>>>(gbuf_f, hf, cf, Hbf_f, enc_f, enc_bf, i, 0);
    gemm(Hbf_b, H, Whhb_bf, H, gbuf_b, 4*H, nullptr, Gxb + (size_t)(S-1-i)*B*4*H, 4*H, B, 4*H, H, 0, 0);
    k_enc_gate<<<dim3((B*H + 255)/256), dim3(256), 0, stream>>>(gbuf_b, hb, cb, Hbf_b, enc_f, enc_bf, S-1-i, 1);
  }

  // attention keys + decoder initial state
  gemm(enc_bf, 2*H, Uabf, 2*H, ua_keys, H, bua, nullptr, 0, B*S, H, 2*H, 0, 0);
  k_cat_states<<<dim3((B*2*H + 255)/256), dim3(256), 0, stream>>>(hf, hb, cf, cb, hcat, ccat);
  gemm(hcat, 2*H, Wrhbf, 2*H, hd, H, brh, nullptr, 0, B, H, 2*H, 0, 0);
  gemm(ccat, 2*H, Wrcbf, 2*H, cd, H, brc, nullptr, 0, B, H, 2*H, 0, 0);
  k_h2xcat<<<dim3((B*H + 255)/256), dim3(256), 0, stream>>>(hd, xcat);

  // ---- decoder recurrence ----
  for (int t = 0; t < T; t++) {
    gemm(xcat + 2*H, 3*H, Wabf, H, qbuf, H, ba, nullptr, 0, B, H, H, 0, 0);        // q = h@Wa.T+ba
    k_attn_scores<<<dim3(B*S/8), dim3(32, 8), 0, stream>>>(qbuf, ua_keys, Va, bva, scores);
    k_attn_ctx<<<dim3(B), dim3(128), 0, stream>>>(scores, enc_f, xcat, out + OFF_AT, t);
    gemm(xcat, 3*H, Wcatbf, 3*H, gbuf_f, 4*H, nullptr, Gemb + (size_t)t*B*4*H, 4*H, B, 4*H, 3*H, 0, 0);
    k_dec_gate<<<dim3((B*H + 255)/256), dim3(256), 0, stream>>>(gbuf_f, hd, cd, xcat, Hdec, t);
  }

  // ---- fused output projection for all steps, then in-place log-softmax ----
  gemm(Hdec, H, Woutbf, H, out, V, bout, nullptr, 0, T*B, V, H, B, T);             // writes [B,T,V]
  k_logsoftmax<<<dim3(B*T), dim3(256), 0, stream>>>(out, V);
  k_copy_states<<<dim3((B*H + 255)/256), dim3(256), 0, stream>>>(hd, cd, out + OFF_HT, out + OFF_CT);
}
